// NeuronGPTOSSBlock_48438641164455
// MI455X (gfx1250) — matrix operand order repaired and checked by reading, not compiled, not run on hardware
//
#include <hip/hip_runtime.h>
#include <math.h>

// ---------------------------------------------------------------------------
// GPT-OSS block for MI455X (gfx1250, wave32, WMMA bf16 16x16x32)
// GEMMs: one wave computes a 32x32 output macro-tile (2x2 WMMA tiles, 4
// independent accumulator chains) to double FLOP/byte and keep the XDL pipe
// fed. Weight streams are prefetched (global_prefetch_b8).
// ---------------------------------------------------------------------------
constexpr int S_   = 2048;
constexpr int H_   = 1024;
constexpr int NH_  = 16;
constexpr int NKV_ = 4;
constexpr int HD_  = 64;
constexpr int I_   = 1024;
constexpr int E_   = 32;
constexpr int TOPK = 4;
constexpr int WIN_ = 128;

typedef __bf16        v16bf __attribute__((ext_vector_type(16)));
typedef float         v8f   __attribute__((ext_vector_type(8)));
typedef unsigned int  u32x4 __attribute__((ext_vector_type(4)));

union BfVec { v16bf v; u32x4 u[2]; };

#define V8F_ZERO {0.f, 0.f, 0.f, 0.f, 0.f, 0.f, 0.f, 0.f}

__device__ __forceinline__ unsigned short f2bf(float f) {
    unsigned int u = __float_as_uint(f);
    u += 0x7FFFu + ((u >> 16) & 1u);          // round-to-nearest-even
    return (unsigned short)(u >> 16);
}

__device__ __forceinline__ u32x4 ld4(const unsigned short* p) {
    return *(const u32x4*)p;                   // 16B vector load
}

// D(16x16,f32) += A(16x32,bf16) x B(32x16,bf16)
// A lane layout (ISA 7.12.2): lanes 0-15 K=0..7,16..23 ; lanes 16-31 K=8..15,24..31
// B lane layout: same K order as A: lanes 0-15 K=0..7,16..23 ; lanes 16-31 K=8..15,24..31
__device__ __forceinline__ v8f wmma_bf16(u32x4 a0, u32x4 a1, u32x4 b0, u32x4 b1, v8f c) {
    BfVec a, b;
    a.u[0] = a0; a.u[1] = a1;
    b.u[0] = b0; b.u[1] = b1;
    return __builtin_amdgcn_wmma_f32_16x16x32_bf16(false, a.v, false, b.v,
                                                   (short)0, c, false, false);
}

// ---------------------------------------------------------------------------
// Weight convert+transpose: fp32 [K][N] (z slices) -> bf16 [N][K]
// ---------------------------------------------------------------------------
__global__ void transpose_f32_bf16(const float* __restrict__ in,
                                   unsigned short* __restrict__ out, int K, int N) {
    __shared__ float tile[32][33];
    const size_t slice = (size_t)K * N;
    const float* ip = in + slice * blockIdx.z;
    unsigned short* op = out + slice * blockIdx.z;
    const int kb = blockIdx.y * 32, nb = blockIdx.x * 32;
    for (int i = threadIdx.y; i < 32; i += 8) {
        int k = kb + i, n = nb + threadIdx.x;
        tile[i][threadIdx.x] = (k < K && n < N) ? ip[(size_t)k * N + n] : 0.f;
    }
    __syncthreads();
    for (int i = threadIdx.y; i < 32; i += 8) {
        int n = nb + i, k = kb + threadIdx.x;
        if (n < N && k < K) op[(size_t)n * K + k] = f2bf(tile[threadIdx.x][i]);
    }
}

// ---------------------------------------------------------------------------
// RMSNorm (H=1024). Writes bf16 (GEMM input) and optionally fp32 (router).
// ---------------------------------------------------------------------------
__global__ void rmsnorm_k(const float* __restrict__ in, const float* __restrict__ w,
                          unsigned short* __restrict__ obf, float* __restrict__ of32) {
    const int t = blockIdx.x;
    __shared__ float red[256];
    const float* row = in + (size_t)t * H_;
    float ss = 0.f;
    for (int i = threadIdx.x; i < H_; i += 256) { float v = row[i]; ss += v * v; }
    red[threadIdx.x] = ss; __syncthreads();
    for (int o = 128; o > 0; o >>= 1) {
        if (threadIdx.x < o) red[threadIdx.x] += red[threadIdx.x + o];
        __syncthreads();
    }
    const float inv = rsqrtf(red[0] / (float)H_ + 1e-5f);
    for (int i = threadIdx.x; i < H_; i += 256) {
        float v = row[i] * inv * w[i];
        obf[(size_t)t * H_ + i] = f2bf(v);
        if (of32) of32[(size_t)t * H_ + i] = v;
    }
}

// ---------------------------------------------------------------------------
// Generic WMMA GEMM: C[M,N] f32 = A_bf16[M,K] @ WT_bf16[N,K]^T (+residual)
// One wave per 32x32 macro-tile; grid = (N/32, M/32).
// ---------------------------------------------------------------------------
__global__ __launch_bounds__(32) void wmma_gemm_f32(
        const unsigned short* __restrict__ A, const unsigned short* __restrict__ WT,
        const float* __restrict__ residual, float* __restrict__ C, int N, int K) {
    const int n0 = blockIdx.x * 32, m0 = blockIdx.y * 32;
    const int lane = threadIdx.x, half = lane >> 4, l = lane & 15;
    const unsigned short* ar0 = A + (size_t)(m0 + l) * K + half * 8;
    const unsigned short* ar1 = ar0 + (size_t)16 * K;
    const unsigned short* br0 = WT + (size_t)(n0 + l) * K + half * 8;
    const unsigned short* br1 = br0 + (size_t)16 * K;
    v8f acc00 = V8F_ZERO, acc01 = V8F_ZERO, acc10 = V8F_ZERO, acc11 = V8F_ZERO;
    for (int k = 0; k < K; k += 32) {
        __builtin_prefetch(br0 + k + 512, 0, 1);   // stream-ahead on weights
        __builtin_prefetch(br1 + k + 512, 0, 1);
        const u32x4 a00 = ld4(ar0 + k), a01 = ld4(ar0 + k + 16);
        const u32x4 a10 = ld4(ar1 + k), a11 = ld4(ar1 + k + 16);
        const u32x4 b00 = ld4(br0 + k), b01 = ld4(br0 + k + 16);
        const u32x4 b10 = ld4(br1 + k), b11 = ld4(br1 + k + 16);
        acc00 = wmma_bf16(a00, a01, b00, b01, acc00);
        acc01 = wmma_bf16(a00, a01, b10, b11, acc01);
        acc10 = wmma_bf16(a10, a11, b00, b01, acc10);
        acc11 = wmma_bf16(a10, a11, b10, b11, acc11);
    }
    const int col = n0 + l, rbase = half * 8;
    for (int r = 0; r < 8; ++r) {
        const int mA = m0 + rbase + r, mB = mA + 16;
        size_t i00 = (size_t)mA * N + col;
        size_t i01 = (size_t)mA * N + col + 16;
        size_t i10 = (size_t)mB * N + col;
        size_t i11 = (size_t)mB * N + col + 16;
        float v00 = acc00[r], v01 = acc01[r], v10 = acc10[r], v11 = acc11[r];
        if (residual) {
            v00 += residual[i00]; v01 += residual[i01];
            v10 += residual[i10]; v11 += residual[i11];
        }
        C[i00] = v00; C[i01] = v01; C[i10] = v10; C[i11] = v11;
    }
}

// ---------------------------------------------------------------------------
// YaRN RoPE, in place on fp32 q [S][NH*HD] and k [S][NKV*HD]
// ---------------------------------------------------------------------------
__global__ void rope_k(float* __restrict__ q, float* __restrict__ k,
                       const int* __restrict__ pos) {
    const int total = S_ * (NH_ + NKV_) * 32;
    int gid = blockIdx.x * blockDim.x + threadIdx.x;
    if (gid >= total) return;
    const int i = gid & 31;
    int rest = gid >> 5;
    const int head = rest % (NH_ + NKV_);
    const int t = rest / (NH_ + NKV_);
    const float log_theta = logf(150000.0f);
    const float TWO_PI = 6.283185307179586f;
    const float low  = 32.0f * logf(4096.0f / (32.0f * TWO_PI)) / log_theta;
    const float high = 32.0f * logf(4096.0f / ( 1.0f * TWO_PI)) / log_theta;
    const float freq = expf(log_theta * (float)i / 32.0f);       // THETA^(i/32)
    float ramp = ((float)i - low) / (high - low);
    float mask = 1.0f - fminf(fmaxf(ramp, 0.f), 1.f);
    float inv  = (1.0f / (32.0f * freq)) * (1.0f - mask) + (1.0f / freq) * mask;
    const float conc = 0.1f * logf(32.0f) + 1.0f;
    const float ang = (float)pos[t] * inv;
    const float c = cosf(ang) * conc, s = sinf(ang) * conc;
    float* ptr = (head < NH_) ? (q + (size_t)t * (NH_ * HD_) + head * HD_)
                              : (k + (size_t)t * (NKV_ * HD_) + (head - NH_) * HD_);
    const float x1 = ptr[i], x2 = ptr[i + 32];
    ptr[i]      = x1 * c - x2 * s;
    ptr[i + 32] = x2 * c + x1 * s;
}

// ---------------------------------------------------------------------------
// Sliding-window attention with sink. Block of 128 threads per (query, head).
// Writes bf16 attn output (GEMM input for O-proj).
// ---------------------------------------------------------------------------
__global__ void attn_k(const float* __restrict__ q, const float* __restrict__ k,
                       const float* __restrict__ v, const int* __restrict__ pos,
                       const float* __restrict__ sinks,
                       unsigned short* __restrict__ obf) {
    const int si = blockIdx.x, h = blockIdx.y, tid = threadIdx.x;
    const int kvh = h >> 2;                               // NH/NKV = 4
    __shared__ float qs[HD_];
    __shared__ float sc[WIN_];
    __shared__ float red[WIN_];
    if (tid < HD_) qs[tid] = q[(size_t)si * (NH_ * HD_) + h * HD_ + tid];
    __syncthreads();
    const int j = si - (WIN_ - 1) + tid;
    float score = -1e9f;
    if (j >= 0) {
        const int d = pos[si] - pos[j];
        if (d >= 0 && d < WIN_) {
            const float* kr = k + (size_t)j * (NKV_ * HD_) + kvh * HD_;
            float a = 0.f;
            #pragma unroll
            for (int dd = 0; dd < HD_; ++dd) a += qs[dd] * kr[dd];
            score = a * 0.125f;                           // 1/sqrt(64)
        }
    }
    sc[tid] = score;
    red[tid] = score;
    __syncthreads();
    for (int o = 64; o > 0; o >>= 1) {
        if (tid < o) red[tid] = fmaxf(red[tid], red[tid + o]);
        __syncthreads();
    }
    const float snk = sinks[h];
    const float mx = fmaxf(red[0], snk);
    __syncthreads();
    const float ex = __expf(score - mx);                  // masked -> 0
    red[tid] = ex;
    __syncthreads();
    for (int o = 64; o > 0; o >>= 1) {
        if (tid < o) red[tid] += red[tid + o];
        __syncthreads();
    }
    const float denom = red[0] + __expf(snk - mx);
    __syncthreads();
    sc[tid] = ex / denom;
    __syncthreads();
    if (tid < HD_) {
        float a = 0.f;
        for (int t2 = 0; t2 < WIN_; ++t2) {
            const int jj = si - (WIN_ - 1) + t2;
            if (jj < 0) continue;
            a += sc[t2] * v[(size_t)jj * (NKV_ * HD_) + kvh * HD_ + tid];
        }
        obf[(size_t)si * (NH_ * HD_) + h * HD_ + tid] = f2bf(a);
    }
}

// ---------------------------------------------------------------------------
// Router: one wave per token, lane e computes logit for expert e; top-4 via
// shuffle argmax; softmax weights; histogram expert counts.
// ---------------------------------------------------------------------------
__global__ __launch_bounds__(32) void router_k(
        const float* __restrict__ x, const float* __restrict__ rw,
        const float* __restrict__ rb, int* __restrict__ tidx,
        float* __restrict__ tw, int* __restrict__ counts) {
    const int t = blockIdx.x, lane = threadIdx.x;
    const float* xr = x + (size_t)t * H_;
    float acc = rb[lane];
    for (int hh = 0; hh < H_; ++hh) acc += xr[hh] * rw[(size_t)hh * E_ + lane];
    float cur = acc;
    float selv[TOPK]; int seli[TOPK];
    for (int jj = 0; jj < TOPK; ++jj) {
        float mv = cur; int mi = lane;
        for (int o = 16; o > 0; o >>= 1) {
            float ov = __shfl_xor(mv, o, 32);
            int   oi = __shfl_xor(mi, o, 32);
            if (ov > mv || (ov == mv && oi < mi)) { mv = ov; mi = oi; }
        }
        selv[jj] = mv; seli[jj] = mi;
        if (lane == mi) cur = -3.0e38f;
    }
    if (lane == 0) {
        float m4 = selv[0], sum = 0.f, e4[TOPK];
        for (int jj = 0; jj < TOPK; ++jj) { e4[jj] = __expf(selv[jj] - m4); sum += e4[jj]; }
        for (int jj = 0; jj < TOPK; ++jj) {
            tidx[t * TOPK + jj] = seli[jj];
            tw[t * TOPK + jj] = e4[jj] / sum;
            atomicAdd(&counts[seli[jj]], 1);
        }
    }
}

__global__ void scan_k(const int* __restrict__ counts, int* __restrict__ offs,
                       int* __restrict__ cursor) {
    if (threadIdx.x == 0 && blockIdx.x == 0) {
        int run = 0;
        for (int e = 0; e < E_; ++e) { offs[e] = run; cursor[e] = run; run += counts[e]; }
    }
}

__global__ void fill_k(const int* __restrict__ tidx, int* __restrict__ cursor,
                       int* __restrict__ perm) {
    const int gid = blockIdx.x * blockDim.x + threadIdx.x;
    if (gid < S_ * TOPK) {
        const int e = tidx[gid];
        const int p = atomicAdd(&cursor[e], 1);
        perm[p] = gid;                       // encodes token*4 + slot
    }
}

// ---------------------------------------------------------------------------
// MoE gate+up fused: 32x32 macro-tile per wave, two weight streams (gate+up)
// sharing the A tiles -> 8 WMMAs per 12 loads. clamp + GLU activation,
// store bf16 hmid at expert-sorted positions.
// ---------------------------------------------------------------------------
__global__ __launch_bounds__(32) void moe_gateup_k(
        const unsigned short* __restrict__ xbf, const unsigned short* __restrict__ wgT,
        const unsigned short* __restrict__ wuT, const float* __restrict__ bg,
        const float* __restrict__ bu, const int* __restrict__ counts,
        const int* __restrict__ offs, const int* __restrict__ perm,
        unsigned short* __restrict__ hmid) {
    const int e = blockIdx.z;
    const int cnt = counts[e];
    const int m0 = blockIdx.y * 32;
    if (m0 >= cnt) return;                               // uniform per wave
    const int n0 = blockIdx.x * 32;
    const int lane = threadIdx.x, half = lane >> 4, l = lane & 15;
    const int seg = offs[e];
    int mr0 = m0 + l;      if (mr0 > cnt - 1) mr0 = cnt - 1;
    int mr1 = m0 + 16 + l; if (mr1 > cnt - 1) mr1 = cnt - 1;
    const int tok0 = perm[seg + mr0] >> 2;
    const int tok1 = perm[seg + mr1] >> 2;
    const unsigned short* ar0 = xbf + (size_t)tok0 * H_ + half * 8;
    const unsigned short* ar1 = xbf + (size_t)tok1 * H_ + half * 8;
    const size_t wb = (size_t)e * I_ * H_ + (size_t)(n0 + l) * H_ + half * 8;
    const unsigned short* gr0 = wgT + wb;
    const unsigned short* gr1 = gr0 + (size_t)16 * H_;
    const unsigned short* ur0 = wuT + wb;
    const unsigned short* ur1 = ur0 + (size_t)16 * H_;
    v8f ag00 = V8F_ZERO, ag01 = V8F_ZERO, ag10 = V8F_ZERO, ag11 = V8F_ZERO;
    v8f au00 = V8F_ZERO, au01 = V8F_ZERO, au10 = V8F_ZERO, au11 = V8F_ZERO;
    for (int kk = 0; kk < H_; kk += 32) {
        __builtin_prefetch(gr0 + kk + 512, 0, 1);
        __builtin_prefetch(ur0 + kk + 512, 0, 1);
        const u32x4 a00 = ld4(ar0 + kk), a01 = ld4(ar0 + kk + 16);
        const u32x4 a10 = ld4(ar1 + kk), a11 = ld4(ar1 + kk + 16);
        const u32x4 g00 = ld4(gr0 + kk), g01 = ld4(gr0 + kk + 16);
        const u32x4 g10 = ld4(gr1 + kk), g11 = ld4(gr1 + kk + 16);
        const u32x4 u00 = ld4(ur0 + kk), u01 = ld4(ur0 + kk + 16);
        const u32x4 u10 = ld4(ur1 + kk), u11 = ld4(ur1 + kk + 16);
        ag00 = wmma_bf16(a00, a01, g00, g01, ag00);
        ag01 = wmma_bf16(a00, a01, g10, g11, ag01);
        ag10 = wmma_bf16(a10, a11, g00, g01, ag10);
        ag11 = wmma_bf16(a10, a11, g10, g11, ag11);
        au00 = wmma_bf16(a00, a01, u00, u01, au00);
        au01 = wmma_bf16(a00, a01, u10, u11, au01);
        au10 = wmma_bf16(a10, a11, u00, u01, au10);
        au11 = wmma_bf16(a10, a11, u10, u11, au11);
    }
    const int col0 = n0 + l, col1 = col0 + 16;
    const float bg0 = bg[e * I_ + col0], bg1 = bg[e * I_ + col1];
    const float bu0 = bu[e * I_ + col0], bu1 = bu[e * I_ + col1];
    const int rbase = half * 8;
    for (int r = 0; r < 8; ++r) {
        #pragma unroll
        for (int mi = 0; mi < 2; ++mi) {
            const int m = m0 + mi * 16 + rbase + r;
            if (m >= cnt) continue;
            const float gA = mi ? ag10[r] : ag00[r];
            const float gB = mi ? ag11[r] : ag01[r];
            const float uA = mi ? au10[r] : au00[r];
            const float uB = mi ? au11[r] : au01[r];
            float g0 = fminf(gA + bg0, 7.0f);
            float g1 = fminf(gB + bg1, 7.0f);
            float u0 = fminf(fmaxf(uA + bu0, -7.0f), 7.0f);
            float u1 = fminf(fmaxf(uB + bu1, -7.0f), 7.0f);
            const float h0 = (u0 + 1.0f) * (g0 / (1.0f + __expf(-1.702f * g0)));
            const float h1 = (u1 + 1.0f) * (g1 / (1.0f + __expf(-1.702f * g1)));
            hmid[(size_t)(seg + m) * I_ + col0] = f2bf(h0);
            hmid[(size_t)(seg + m) * I_ + col1] = f2bf(h1);
        }
    }
}

// ---------------------------------------------------------------------------
// MoE down projection: 32x32 macro-tile per wave. hmid @ wd_e + bd_e, scaled
// by routing weight, scattered into per-(token,slot) contrib buffer
// (deterministic final sum regardless of atomic grouping order).
// ---------------------------------------------------------------------------
__global__ __launch_bounds__(32) void moe_down_k(
        const unsigned short* __restrict__ hmid, const unsigned short* __restrict__ wdT,
        const float* __restrict__ bd, const float* __restrict__ tw,
        const int* __restrict__ counts, const int* __restrict__ offs,
        const int* __restrict__ perm, float* __restrict__ contrib) {
    const int e = blockIdx.z;
    const int cnt = counts[e];
    const int m0 = blockIdx.y * 32;
    if (m0 >= cnt) return;
    const int n0 = blockIdx.x * 32;
    const int lane = threadIdx.x, half = lane >> 4, l = lane & 15;
    const int seg = offs[e];
    int mr0 = m0 + l;      if (mr0 > cnt - 1) mr0 = cnt - 1;
    int mr1 = m0 + 16 + l; if (mr1 > cnt - 1) mr1 = cnt - 1;
    const unsigned short* ar0 = hmid + (size_t)(seg + mr0) * I_ + half * 8;
    const unsigned short* ar1 = hmid + (size_t)(seg + mr1) * I_ + half * 8;
    const unsigned short* br0 = wdT + (size_t)e * H_ * I_ + (size_t)(n0 + l) * I_ + half * 8;
    const unsigned short* br1 = br0 + (size_t)16 * I_;
    v8f acc00 = V8F_ZERO, acc01 = V8F_ZERO, acc10 = V8F_ZERO, acc11 = V8F_ZERO;
    for (int kk = 0; kk < I_; kk += 32) {
        __builtin_prefetch(br0 + kk + 512, 0, 1);
        __builtin_prefetch(br1 + kk + 512, 0, 1);
        const u32x4 a00 = ld4(ar0 + kk), a01 = ld4(ar0 + kk + 16);
        const u32x4 a10 = ld4(ar1 + kk), a11 = ld4(ar1 + kk + 16);
        const u32x4 b00 = ld4(br0 + kk), b01 = ld4(br0 + kk + 16);
        const u32x4 b10 = ld4(br1 + kk), b11 = ld4(br1 + kk + 16);
        acc00 = wmma_bf16(a00, a01, b00, b01, acc00);
        acc01 = wmma_bf16(a00, a01, b10, b11, acc01);
        acc10 = wmma_bf16(a10, a11, b00, b01, acc10);
        acc11 = wmma_bf16(a10, a11, b10, b11, acc11);
    }
    const int col0 = n0 + l, col1 = col0 + 16;
    const float bd0 = bd[e * H_ + col0], bd1 = bd[e * H_ + col1];
    const int rbase = half * 8;
    for (int r = 0; r < 8; ++r) {
        #pragma unroll
        for (int mi = 0; mi < 2; ++mi) {
            const int m = m0 + mi * 16 + rbase + r;
            if (m >= cnt) continue;
            const int pe = perm[seg + m];                // token*4 + slot
            const float w = tw[pe];
            const float vA = mi ? acc10[r] : acc00[r];
            const float vB = mi ? acc11[r] : acc01[r];
            contrib[(size_t)pe * H_ + col0] = w * (vA + bd0);
            contrib[(size_t)pe * H_ + col1] = w * (vB + bd1);
        }
    }
}

__global__ void final_k(const float* __restrict__ h2, const float* __restrict__ contrib,
                        float* __restrict__ out) {
    const int gid = blockIdx.x * blockDim.x + threadIdx.x;
    if (gid >= S_ * H_) return;
    const int t = gid >> 10, hh = gid & (H_ - 1);
    float s = h2[gid];
    #pragma unroll
    for (int jj = 0; jj < TOPK; ++jj)
        s += contrib[(size_t)(t * TOPK + jj) * H_ + hh];
    out[gid] = s;
}

// ---------------------------------------------------------------------------
extern "C" void kernel_launch(void* const* d_in, const int* in_sizes, int n_in,
                              void* d_out, int out_size, void* d_ws, size_t ws_size,
                              hipStream_t stream) {
    const float* hidden = (const float*)d_in[0];
    const int*   pos    = (const int*)d_in[1];
    const float* ln1w   = (const float*)d_in[2];
    const float* ln2w   = (const float*)d_in[3];
    const float* wq     = (const float*)d_in[4];
    const float* wk     = (const float*)d_in[5];
    const float* wv     = (const float*)d_in[6];
    const float* wo     = (const float*)d_in[7];
    const float* sinks  = (const float*)d_in[8];
    const float* rw     = (const float*)d_in[9];
    const float* rb     = (const float*)d_in[10];
    const float* wg     = (const float*)d_in[11];
    const float* bg     = (const float*)d_in[12];
    const float* wu     = (const float*)d_in[13];
    const float* bu     = (const float*)d_in[14];
    const float* wd     = (const float*)d_in[15];
    const float* bd     = (const float*)d_in[16];
    float* out = (float*)d_out;

    // ---- workspace bump allocator (256B aligned) ----
    char* p = (char*)d_ws;
    auto alloc = [&](size_t bytes) -> char* {
        char* r = p;
        p += (bytes + 255) & ~(size_t)255;
        return r;
    };
    unsigned short* WQT  = (unsigned short*)alloc((size_t)H_ * H_ * 2);
    unsigned short* WKT  = (unsigned short*)alloc((size_t)(NKV_ * HD_) * H_ * 2);
    unsigned short* WVT  = (unsigned short*)alloc((size_t)(NKV_ * HD_) * H_ * 2);
    unsigned short* WOT  = (unsigned short*)alloc((size_t)H_ * H_ * 2);
    unsigned short* WGT  = (unsigned short*)alloc((size_t)E_ * I_ * H_ * 2);
    unsigned short* WUT  = (unsigned short*)alloc((size_t)E_ * I_ * H_ * 2);
    unsigned short* WDT  = (unsigned short*)alloc((size_t)E_ * H_ * I_ * 2);
    unsigned short* X1   = (unsigned short*)alloc((size_t)S_ * H_ * 2);
    unsigned short* X2BF = (unsigned short*)alloc((size_t)S_ * H_ * 2);
    float* X2F    = (float*)alloc((size_t)S_ * H_ * 4);
    float* Q      = (float*)alloc((size_t)S_ * NH_ * HD_ * 4);
    float* KB     = (float*)alloc((size_t)S_ * NKV_ * HD_ * 4);
    float* VB     = (float*)alloc((size_t)S_ * NKV_ * HD_ * 4);
    unsigned short* ATTN = (unsigned short*)alloc((size_t)S_ * H_ * 2);
    float* H2     = (float*)alloc((size_t)S_ * H_ * 4);
    unsigned short* HMID = (unsigned short*)alloc((size_t)S_ * TOPK * I_ * 2);
    float* CONTRIB = (float*)alloc((size_t)S_ * TOPK * H_ * 4);
    int*   TIDX   = (int*)alloc(S_ * TOPK * 4);
    float* TW     = (float*)alloc(S_ * TOPK * 4);
    int*   COUNTS = (int*)alloc(E_ * 4);
    int*   OFFS   = (int*)alloc(E_ * 4);
    int*   CURSOR = (int*)alloc(E_ * 4);
    int*   PERM   = (int*)alloc(S_ * TOPK * 4);

    hipMemsetAsync(COUNTS, 0, E_ * sizeof(int), stream);

    // ---- weight convert + transpose to bf16 [N][K] ----
    dim3 tb(32, 8);
    transpose_f32_bf16<<<dim3(32, 32, 1),  tb, 0, stream>>>(wq, WQT, H_, H_);
    transpose_f32_bf16<<<dim3(8,  32, 1),  tb, 0, stream>>>(wk, WKT, H_, NKV_ * HD_);
    transpose_f32_bf16<<<dim3(8,  32, 1),  tb, 0, stream>>>(wv, WVT, H_, NKV_ * HD_);
    transpose_f32_bf16<<<dim3(32, 32, 1),  tb, 0, stream>>>(wo, WOT, H_, H_);
    transpose_f32_bf16<<<dim3(32, 32, E_), tb, 0, stream>>>(wg, WGT, H_, I_);
    transpose_f32_bf16<<<dim3(32, 32, E_), tb, 0, stream>>>(wu, WUT, H_, I_);
    transpose_f32_bf16<<<dim3(32, 32, E_), tb, 0, stream>>>(wd, WDT, I_, H_);

    // ---- attention path ----
    rmsnorm_k<<<S_, 256, 0, stream>>>(hidden, ln1w, X1, nullptr);
    wmma_gemm_f32<<<dim3(H_ / 32, S_ / 32), 32, 0, stream>>>(X1, WQT, nullptr, Q, H_, H_);
    wmma_gemm_f32<<<dim3((NKV_ * HD_) / 32, S_ / 32), 32, 0, stream>>>(X1, WKT, nullptr, KB, NKV_ * HD_, H_);
    wmma_gemm_f32<<<dim3((NKV_ * HD_) / 32, S_ / 32), 32, 0, stream>>>(X1, WVT, nullptr, VB, NKV_ * HD_, H_);
    {
        const int total = S_ * (NH_ + NKV_) * 32;
        rope_k<<<(total + 255) / 256, 256, 0, stream>>>(Q, KB, pos);
    }
    attn_k<<<dim3(S_, NH_), WIN_, 0, stream>>>(Q, KB, VB, pos, sinks, ATTN);
    wmma_gemm_f32<<<dim3(H_ / 32, S_ / 32), 32, 0, stream>>>(ATTN, WOT, hidden, H2, H_, H_);

    // ---- router ----
    rmsnorm_k<<<S_, 256, 0, stream>>>(H2, ln2w, X2BF, X2F);
    router_k<<<S_, 32, 0, stream>>>(X2F, rw, rb, TIDX, TW, COUNTS);
    scan_k<<<1, 32, 0, stream>>>(COUNTS, OFFS, CURSOR);
    fill_k<<<(S_ * TOPK + 255) / 256, 256, 0, stream>>>(TIDX, CURSOR, PERM);

    // ---- expert-grouped MoE GEMMs ----
    moe_gateup_k<<<dim3(I_ / 32, S_ / 32, E_), 32, 0, stream>>>(
        X2BF, WGT, WUT, bg, bu, COUNTS, OFFS, PERM, HMID);
    moe_down_k<<<dim3(H_ / 32, S_ / 32, E_), 32, 0, stream>>>(
        HMID, WDT, bd, TW, COUNTS, OFFS, PERM, CONTRIB);

    // ---- final residual + weighted expert sum ----
    final_k<<<(S_ * H_ + 255) / 256, 256, 0, stream>>>(H2, CONTRIB, out);
}
